// SelfAttention_34462817583144
// MI455X (gfx1250) — compile-verified, matrix operand order checked by reading, not run on hardware
//
#include <hip/hip_runtime.h>

typedef __bf16 bf16;
typedef __attribute__((ext_vector_type(16))) __bf16 v16bf;
typedef __attribute__((ext_vector_type(8)))  float  v8f;
typedef __attribute__((address_space(3)))    bf16   lds_bf16;

union FragBF { v16bf v; uint4 u[2]; };
union Half8  { uint4 u; bf16 h[8]; };

__device__ inline v8f zero8() {
  v8f z;
  #pragma unroll
  for (int i = 0; i < 8; i++) z[i] = 0.f;
  return z;
}

// Async copy of 16 bytes global -> LDS (GLOBAL_LOAD_ASYNC_TO_LDS_B128).
// Tracked by ASYNCcnt; completed by s_wait_asynccnt before the barrier.
__device__ inline unsigned lds_off(const bf16* p) {
  return (unsigned)(size_t)(lds_bf16*)p;   // addrspacecast -> LDS byte offset
}
__device__ inline void async_cp16(unsigned dst_lds, const bf16* src) {
  asm volatile("global_load_async_to_lds_b128 %0, %1, off"
               :: "v"(dst_lds), "v"(src) : "memory");
}
__device__ inline void async_wait0() {
  asm volatile("s_wait_asynccnt 0x0" ::: "memory");
}

// Load a 16x32 bf16 fragment (A or B operand) from an LDS tile stored as
// [row][k] with row stride `ld` halves (ld must be a multiple of 8).
// Matches CDNA5 16-bit A-matrix layout: lane = row (mod 16),
// lanes 0-15 hold K {0..7,16..23}, lanes 16-31 hold K {8..15,24..31}.
__device__ inline v16bf ld_frag(const bf16* base, int row0, int k0, int ld) {
  const int lane = threadIdx.x & 31;
  const int r  = row0 + (lane & 15);
  const int kb = k0 + ((lane & 16) ? 8 : 0);
  FragBF f;
  f.u[0] = *(const uint4*)(base + r * ld + kb);
  f.u[1] = *(const uint4*)(base + r * ld + kb + 16);
  return f.v;
}

__global__ __launch_bounds__(256) void cvt_f32_bf16(
    const float* __restrict__ in, bf16* __restrict__ out, size_t n) {
  size_t i = (size_t)blockIdx.x * 256 + threadIdx.x;
  if (i < n) out[i] = (bf16)in[i];
}

constexpr int GLDT = 40;  // GEMM LDS tile stride in halves (80B, 16B-aligned)

// One 32-deep K step of the block-level MMA: wave computes its 32x64 patch.
__device__ inline void mma_step(const bf16* As, const bf16* Bs,
                                int wm, int wn, v8f acc[2][4]) {
  v16bf a0 = ld_frag(As, wm * 32 + 0,  0, GLDT);
  v16bf a1 = ld_frag(As, wm * 32 + 16, 0, GLDT);
  #pragma unroll
  for (int nt = 0; nt < 4; nt++) {
    v16bf b = ld_frag(Bs, wn * 64 + nt * 16, 0, GLDT);
    acc[0][nt] = __builtin_amdgcn_wmma_f32_16x16x32_bf16(
        false, a0, false, b, (short)0, acc[0][nt], false, false);
    acc[1][nt] = __builtin_amdgcn_wmma_f32_16x16x32_bf16(
        false, a1, false, b, (short)0, acc[1][nt], false, false);
  }
}

// C[m,n] = sum_k A[m,k] * W[n,k]   (A: MxK row-major, W: NxK row-major)
// BM=BN=128, BK=32; 256 threads = 8 waves; wave -> 32x64 patch (2x4 tiles).
// Double-buffered LDS with async global->LDS staging: stage tile kt+1 via
// ASYNCcnt-tracked DMA while WMMAs consume tile kt; one barrier per K step;
// tile kt+2 prefetched (global_prefetch_b8). NK = K/32 must be even.
template <bool OUT_BF16>
__global__ __launch_bounds__(256) void gemm_wmma_bf16(
    const bf16* __restrict__ A, const bf16* __restrict__ W,
    void* __restrict__ C, int M, int N, int K) {
  constexpr int BM = 128, BN = 128, BK = 32, LDT = GLDT;
  __shared__ bf16 As[2][BM * LDT];
  __shared__ bf16 Bs[2][BN * LDT];

  const int tid  = threadIdx.x;
  const int lane = tid & 31;
  const int wave = tid >> 5;
  const int wm = wave >> 1;   // 0..3  -> 32-row band
  const int wn = wave & 1;    // 0..1  -> 64-col band
  const int bRow = blockIdx.y * BM;
  const int bCol = blockIdx.x * BN;

  v8f acc[2][4];
  #pragma unroll
  for (int i = 0; i < 2; i++)
    #pragma unroll
    for (int j = 0; j < 4; j++) acc[i][j] = zero8();

  // each thread copies 16 halves of A-tile and 16 halves of W-tile per step
  const int ldr = (tid * 16) >> 5;     // 0..127
  const int ldc = (tid * 16) & 31;     // 0 or 16
  const size_t aBase = (size_t)(bRow + ldr) * K + ldc;
  const size_t wBase = (size_t)(bCol + ldr) * K + ldc;
  const int lo = ldr * LDT + ldc;
  const unsigned dA0 = lds_off(&As[0][lo]), dA0b = lds_off(&As[0][lo + 8]);
  const unsigned dA1 = lds_off(&As[1][lo]), dA1b = lds_off(&As[1][lo + 8]);
  const unsigned dB0 = lds_off(&Bs[0][lo]), dB0b = lds_off(&Bs[0][lo + 8]);
  const unsigned dB1 = lds_off(&Bs[1][lo]), dB1b = lds_off(&Bs[1][lo + 8]);
  const int NK = K / BK;               // even (K = 4096)

  // preload tile 0 into buffer 0
  async_cp16(dA0,  A + aBase);
  async_cp16(dA0b, A + aBase + 8);
  async_cp16(dB0,  W + wBase);
  async_cp16(dB0b, W + wBase + 8);
  async_wait0();
  __syncthreads();

  #pragma unroll 1
  for (int kt = 0; kt < NK; kt += 2) {
    { // stage tile kt+1 into buffer 1 (kt+1 <= NK-1 always here)
      const size_t off = (size_t)(kt + 1) * BK;
      async_cp16(dA1,  A + aBase + off);
      async_cp16(dA1b, A + aBase + off + 8);
      async_cp16(dB1,  W + wBase + off);
      async_cp16(dB1b, W + wBase + off + 8);
    }
    if (kt + 2 < NK) {
      const size_t poff = (size_t)(kt + 2) * BK;
      __builtin_prefetch(A + aBase + poff, 0, 0);
      __builtin_prefetch(W + wBase + poff, 0, 0);
    }
    mma_step(As[0], Bs[0], wm, wn, acc);
    async_wait0();
    __syncthreads();

    if (kt + 2 < NK) { // stage tile kt+2 into buffer 0
      const size_t off = (size_t)(kt + 2) * BK;
      async_cp16(dA0,  A + aBase + off);
      async_cp16(dA0b, A + aBase + off + 8);
      async_cp16(dB0,  W + wBase + off);
      async_cp16(dB0b, W + wBase + off + 8);
    }
    if (kt + 3 < NK) {
      const size_t poff = (size_t)(kt + 3) * BK;
      __builtin_prefetch(A + aBase + poff, 0, 0);
      __builtin_prefetch(W + wBase + poff, 0, 0);
    }
    mma_step(As[1], Bs[1], wm, wn, acc);
    async_wait0();
    __syncthreads();
  }

  // C/D layout: VGPR i -> M = i + (lane>=16 ? 8 : 0); N = lane & 15
  const int rhi = (lane >> 4) << 3;
  const int clo = lane & 15;
  #pragma unroll
  for (int mt = 0; mt < 2; mt++)
    #pragma unroll
    for (int nt = 0; nt < 4; nt++) {
      const int row = bRow + wm * 32 + mt * 16 + rhi;
      const int col = bCol + wn * 64 + nt * 16 + clo;
      #pragma unroll
      for (int i = 0; i < 8; i++) {
        float v = acc[mt][nt][i];
        if (OUT_BF16) ((bf16*)C)[(size_t)(row + i) * N + col] = (bf16)v;
        else          ((float*)C)[(size_t)(row + i) * N + col] = v;
      }
    }
}

// RoPE in-place on Q and K buffers laid out [token(2048)][h*128 + d]
__global__ __launch_bounds__(256) void rope_kernel(
    bf16* __restrict__ Qb, bf16* __restrict__ Kb,
    const float* __restrict__ Cs, const float* __restrict__ Sn) {
  const size_t idx = (size_t)blockIdx.x * 256 + threadIdx.x; // pair index
  const int    p    = (int)(idx & 63);
  const size_t rest = idx >> 6;
  const int    hh   = (int)(rest & 31);
  const size_t tok  = rest >> 5;
  const int    s    = (int)(tok & 1023);
  const size_t base = tok * 4096 + (size_t)hh * 128 + 2 * p;
  const float c = Cs[s * 64 + p], sv = Sn[s * 64 + p];
  float qr = (float)Qb[base], qi = (float)Qb[base + 1];
  Qb[base]     = (bf16)(qr * c - qi * sv);
  Qb[base + 1] = (bf16)(qr * sv + qi * c);
  float kr = (float)Kb[base], ki = (float)Kb[base + 1];
  Kb[base]     = (bf16)(kr * c - ki * sv);
  Kb[base + 1] = (bf16)(kr * sv + ki * c);
}

// Flash-style causal attention. Block = 128 thr (4 waves), one (b,h,32-q-rows).
__global__ __launch_bounds__(128) void attn_wmma(
    const bf16* __restrict__ Q, const bf16* __restrict__ Kc,
    const bf16* __restrict__ V, bf16* __restrict__ O) {
  constexpr int S = 1024, HD = 128, QT = 32, KT = 64;
  constexpr int LQ = 136, LK = 136, LV = 72, LP = 72, LS = 65;
  __shared__ bf16  Qs[QT * LQ];
  __shared__ bf16  Ks[KT * LK];
  __shared__ bf16  Vt[HD * LV];   // V transposed: [d][key]
  __shared__ float Sb[QT * LS];
  __shared__ bf16  Pb[QT * LP];
  __shared__ float rowf[QT];

  const int tid  = threadIdx.x;
  const int lane = tid & 31;
  const int wave = tid >> 5;
  const int wm = wave >> 1, wn = wave & 1;
  const int q0 = blockIdx.x * QT;
  const int h  = blockIdx.y;
  const int b  = blockIdx.z;
  const size_t tok0 = (size_t)b * S;
  const int rhi = (lane >> 4) << 3;
  const int clo = lane & 15;

  { // stage Q tile once
    const int r = (tid * 32) >> 7;
    const int c = (tid * 32) & 127;
    const bf16* src = Q + (tok0 + q0 + r) * 4096 + (size_t)h * HD + c;
    #pragma unroll
    for (int j = 0; j < 4; j++)
      *(uint4*)&Qs[r * LQ + c + 8 * j] = *(const uint4*)(src + 8 * j);
  }

  v8f o[4];
  #pragma unroll
  for (int nt = 0; nt < 4; nt++) o[nt] = zero8();
  float m_i = -__builtin_inff(), l_i = 0.f;

  const int nkt = (q0 + QT - 1) / KT + 1;
  for (int kt = 0; kt < nkt; kt++) {
    { // stage K tile [key][d]
      const int r = (tid * 64) >> 7;
      const int c = (tid * 64) & 127;
      const bf16* src = Kc + (tok0 + kt * KT + r) * 4096 + (size_t)h * HD + c;
      #pragma unroll
      for (int j = 0; j < 8; j++)
        *(uint4*)&Ks[r * LK + c + 8 * j] = *(const uint4*)(src + 8 * j);
    }
    { // stage V transposed -> Vt[d][key]
      #pragma unroll
      for (int i = 0; i < 8; i++) {
        const int id  = tid * 8 + i;
        const int key = id >> 4;
        const int db  = (id & 15) * 8;
        Half8 hv;
        hv.u = *(const uint4*)(V + (tok0 + kt * KT + key) * 4096 +
                               (size_t)h * HD + db);
        #pragma unroll
        for (int j = 0; j < 8; j++) Vt[(db + j) * LV + key] = hv.h[j];
      }
    }
    __syncthreads();

    // S-tile = Q * K^T  (wave -> 16x32 patch of the 32x64 S tile)
    v8f sc[2];
    sc[0] = zero8(); sc[1] = zero8();
    #pragma unroll
    for (int kc = 0; kc < 4; kc++) {
      v16bf a = ld_frag(Qs, wm * 16, kc * 32, LQ);
      #pragma unroll
      for (int nt = 0; nt < 2; nt++) {
        v16bf bb = ld_frag(Ks, wn * 32 + nt * 16, kc * 32, LK);
        sc[nt] = __builtin_amdgcn_wmma_f32_16x16x32_bf16(
            false, a, false, bb, (short)0, sc[nt], false, false);
      }
    }
    const float scl = 0.08838834764831845f;  // 1/sqrt(128)
    #pragma unroll
    for (int nt = 0; nt < 2; nt++)
      #pragma unroll
      for (int i = 0; i < 8; i++) {
        const int rl = wm * 16 + rhi + i;
        const int cl = wn * 32 + nt * 16 + clo;
        float v = sc[nt][i];
        if (kt * KT + cl > q0 + rl) v += -1.0e9f;  // causal mask
        Sb[rl * LS + cl] = v * scl;
      }
    __syncthreads();

    // online softmax, one thread per q-row
    if (tid < QT) {
      float tmax = -__builtin_inff();
      for (int j = 0; j < KT; j++) tmax = fmaxf(tmax, Sb[tid * LS + j]);
      const float mnew = fmaxf(m_i, tmax);
      const float rs = __expf(m_i - mnew);
      float sum = 0.f;
      for (int j = 0; j < KT; j++) {
        float p = __expf(Sb[tid * LS + j] - mnew);
        Pb[tid * LP + j] = (bf16)p;
        sum += p;
      }
      l_i = l_i * rs + sum;
      m_i = mnew;
      rowf[tid] = rs;
    }
    __syncthreads();

    // O = O*rescale + P*V   (wave -> 16x64 patch of 32x128 O tile)
    #pragma unroll
    for (int nt = 0; nt < 4; nt++)
      #pragma unroll
      for (int i = 0; i < 8; i++) o[nt][i] *= rowf[wm * 16 + rhi + i];
    #pragma unroll
    for (int kc = 0; kc < 2; kc++) {
      v16bf a = ld_frag(Pb, wm * 16, kc * 32, LP);
      #pragma unroll
      for (int nt = 0; nt < 4; nt++) {
        v16bf bb = ld_frag(Vt, wn * 64 + nt * 16, kc * 32, LV);
        o[nt] = __builtin_amdgcn_wmma_f32_16x16x32_bf16(
            false, a, false, bb, (short)0, o[nt], false, false);
      }
    }
    __syncthreads();
  }

  if (tid < QT) rowf[tid] = 1.f / l_i;
  __syncthreads();
  #pragma unroll
  for (int nt = 0; nt < 4; nt++)
    #pragma unroll
    for (int i = 0; i < 8; i++) {
      const int rl = wm * 16 + rhi + i;
      const int cl = wn * 64 + nt * 16 + clo;
      O[(tok0 + q0 + rl) * 4096 + (size_t)h * HD + cl] =
          (bf16)(o[nt][i] * rowf[rl]);
    }
}

extern "C" void kernel_launch(void* const* d_in, const int* in_sizes, int n_in,
                              void* d_out, int out_size, void* d_ws, size_t ws_size,
                              hipStream_t stream) {
  // inputs: x, start_pos, freqs_cos, freqs_sin, mask, cache_k, cache_v, wq, wk, wv, wo
  const float* x  = (const float*)d_in[0];
  const float* fc = (const float*)d_in[2];
  const float* fs = (const float*)d_in[3];
  const float* wq = (const float*)d_in[7];
  const float* wk = (const float*)d_in[8];
  const float* wv = (const float*)d_in[9];
  const float* wo = (const float*)d_in[10];
  float* out = (float*)d_out;

  const size_t NX = 2ull * 1024 * 4096;   // 8,388,608  activations
  const size_t NW = 4096ull * 4096;       // 16,777,216 weight elements

  char* ws = (char*)d_ws;
  bf16* xb  = (bf16*)ws;                  // NX
  bf16* wqb = (bf16*)(ws + 2 * NX);       // NW each
  bf16* wkb = wqb + NW;
  bf16* wvb = wkb + NW;
  bf16* wob = wvb + NW;
  bf16* qb  = wob + NW;                   // NX each
  bf16* kb  = qb + NX;
  bf16* vb  = kb + NX;
  bf16* ab  = vb + NX;

  cvt_f32_bf16<<<(unsigned)((NX + 255) / 256), 256, 0, stream>>>(x,  xb,  NX);
  cvt_f32_bf16<<<(unsigned)((NW + 255) / 256), 256, 0, stream>>>(wq, wqb, NW);
  cvt_f32_bf16<<<(unsigned)((NW + 255) / 256), 256, 0, stream>>>(wk, wkb, NW);
  cvt_f32_bf16<<<(unsigned)((NW + 255) / 256), 256, 0, stream>>>(wv, wvb, NW);
  cvt_f32_bf16<<<(unsigned)((NW + 255) / 256), 256, 0, stream>>>(wo, wob, NW);

  dim3 gg(4096 / 128, 2048 / 128);
  gemm_wmma_bf16<true ><<<gg, 256, 0, stream>>>(xb, wqb, qb, 2048, 4096, 4096);
  gemm_wmma_bf16<true ><<<gg, 256, 0, stream>>>(xb, wkb, kb, 2048, 4096, 4096);
  gemm_wmma_bf16<true ><<<gg, 256, 0, stream>>>(xb, wvb, vb, 2048, 4096, 4096);

  rope_kernel<<<(unsigned)((2ull * 1024 * 32 * 64) / 256), 256, 0, stream>>>(
      qb, kb, fc, fs);

  dim3 ga(1024 / 32, 32, 2);
  attn_wmma<<<ga, 128, 0, stream>>>(qb, kb, vb, ab);

  gemm_wmma_bf16<false><<<gg, 256, 0, stream>>>(ab, wob, out, 2048, 4096, 4096);
}
